// Causal_RCNN_46858093199881
// MI455X (gfx1250) — matrix operand order repaired and checked by reading, not compiled, not run on hardware
//
#include <hip/hip_runtime.h>
#include <hip/hip_bf16.h>
#include <stdint.h>

typedef __attribute__((ext_vector_type(16))) __bf16 bf16x16;
typedef __attribute__((ext_vector_type(8)))  float  v8f;

#define B_  16
#define L_  1024
#define D_  1024
#define K_  3
#define LB_ (L_ * B_)          // 16384 token rows, m = l*16 + b

union FragU { uint4 q[2]; bf16x16 v; };

// -------- CDNA5 async global->LDS copy (ASYNCcnt path, no VGPR round trip) --------
__device__ __forceinline__ void async_copy_b128(const void* gsrc, void* ldst) {
  // VGLOBAL async-to-LDS: VDST = per-lane LDS byte offset, VADDR = 64-bit global addr.
  // Low 32 bits of a generic pointer to __shared__ are the LDS byte offset.
  uint32_t loff  = (uint32_t)(uintptr_t)ldst;
  uint64_t gaddr = (uint64_t)(uintptr_t)gsrc;
  asm volatile("global_load_async_to_lds_b128 %0, %1, off"
               :: "v"(loff), "v"(gaddr)
               : "memory");
}

__device__ __forceinline__ void wait_async0() {
#if __has_builtin(__builtin_amdgcn_s_wait_asynccnt)
  __builtin_amdgcn_s_wait_asynccnt(0);
#else
  asm volatile("s_wait_asynccnt 0x0" ::: "memory");
#endif
}

// ---------------------------------------------------------------- pack kernels
__global__ __launch_bounds__(256) void pack_x_kernel(const float* __restrict__ x,
                                                     __bf16* __restrict__ xb,
                                                     uint4* __restrict__ zpad) {
  if (blockIdx.x == 0 && threadIdx.x < 64)      // re-zero the causal pad every launch
    zpad[threadIdx.x] = make_uint4(0u, 0u, 0u, 0u);
  size_t i = ((size_t)blockIdx.x * 256 + threadIdx.x) * 4;
  float4 v = *(const float4*)(x + i);
  xb[i + 0] = (__bf16)v.x; xb[i + 1] = (__bf16)v.y;
  xb[i + 2] = (__bf16)v.z; xb[i + 3] = (__bf16)v.w;
}

// wb[dout][k*D + din] = conv_w[dout][din][k]
__global__ __launch_bounds__(256) void pack_wc_kernel(const float* __restrict__ w,
                                                      __bf16* __restrict__ wb) {
  int i = blockIdx.x * 256 + threadIdx.x;          // 0 .. D*3072-1
  int dout = i / (K_ * D_);
  int r    = i - dout * (K_ * D_);
  int k    = r >> 10;
  int din  = r & (D_ - 1);
  wb[i] = (__bf16)w[((size_t)dout * D_ + din) * K_ + k];
}

// wb[e][d] = sru_w[d][e]   (e in [0,3D))
__global__ __launch_bounds__(256) void pack_ws_kernel(const float* __restrict__ w,
                                                      __bf16* __restrict__ wb) {
  int i = blockIdx.x * 256 + threadIdx.x;          // 0 .. 3D*D-1
  int e = i >> 10;
  int d = i & (D_ - 1);
  wb[i] = (__bf16)w[(size_t)d * (3 * D_) + e];
}

// ---------------------------------------------------------------- WMMA GEMM
// MODE 0: cnn_out[m][n] = sum_{k,din} x[b][l-2+k][din] * Wc[n][k*D+din] + conv_b[n]
// MODE 1: U[m][e]       = sum_d normed_bf[m][d] * Ws[e][d]
// 128x128 workgroup tile, 8 waves of 32(M)x64(N), double-buffered LDS filled by
// global_load_async_to_lds_b128, one barrier per K-step of 32.
template <int MODE>
__global__ __launch_bounds__(256) void gemm_wmma_kernel(
    const __bf16* __restrict__ Amat,
    const __bf16* __restrict__ Bw,
    const float*  __restrict__ bias,
    const __bf16* __restrict__ zpad,
    float* __restrict__ Cout) {
  constexpr int KD = (MODE == 0) ? (K_ * D_) : D_;
  constexpr int ND = (MODE == 0) ? D_ : (3 * D_);

  // [buf 0: A(512) B(512)] [buf 1: A(512) B(512)]  (uint4 units, 32 KB total)
  __shared__ uint4 lds[2048];

  const int tid   = threadIdx.x;
  const int lane  = tid & 31;
  const int wv    = tid >> 5;          // 8 waves
  const int wm    = (wv & 3) * 32;     // wave M offset in 128 tile
  const int wn    = (wv >> 2) * 64;    // wave N offset in 128 tile
  const int mbase = blockIdx.y * 128;
  const int nbase = blockIdx.x * 128;
  const int lm    = lane & 15;
  const int lk    = lane >> 4;

  const v8f vzero = {0.f, 0.f, 0.f, 0.f, 0.f, 0.f, 0.f, 0.f};
  v8f acc[2][4];
#pragma unroll
  for (int i = 0; i < 2; ++i)
#pragma unroll
    for (int j = 0; j < 4; ++j) acc[i][j] = vzero;

  // issue 4 async b128 copies per thread for one (A,B) K-tile into buffer `buf`
  auto stage = [&](int buf, int kt) {
    uint4* As = lds + buf * 1024;
    uint4* Bs = As + 512;
#pragma unroll
    for (int s = 0; s < 2; ++s) {
      int idx  = tid + s * 256;        // 512 16B chunks per matrix tile
      int row  = idx >> 2;
      int quad = idx & 3;
      const __bf16* agp;
      if constexpr (MODE == 0) {
        int m = mbase + row, l = m >> 4, b = m & 15;
        int kseg = kt >> 10;                     // conv tap (32-chunk never crosses taps)
        int din  = (kt & (D_ - 1)) + quad * 8;
        int t    = l - 2 + kseg;
        // causal zero padding via address select (EXEC stays all-ones)
        agp = (t >= 0) ? (Amat + ((size_t)(b * L_ + t) * D_ + din))
                       : (zpad + (size_t)(idx & 63) * 8);
      } else {
        agp = Amat + ((size_t)(mbase + row) * KD + kt + quad * 8);
      }
      async_copy_b128(agp, As + idx);
      const __bf16* bgp = Bw + ((size_t)(nbase + row) * KD + kt + quad * 8);
      async_copy_b128(bgp, Bs + idx);
    }
  };

  constexpr int NIT = KD / 32;
  stage(0, 0);

  for (int it = 0; it < NIT; ++it) {
    const int buf = it & 1;
    wait_async0();        // our async copies into `buf` have landed
    __syncthreads();      // everyone's copies landed; everyone done reading buf^1
    if (it + 1 < NIT) stage(buf ^ 1, (it + 1) * 32);   // overlap with compute below

    const uint4* AsQ = lds + buf * 1024;
    const uint4* BsQ = AsQ + 512;

    // A fragment: lane holds row M=lm, K = lk*8..+7 and 16+lk*8..+7  (ISA 16-bit A layout)
    FragU af[2];
#pragma unroll
    for (int i = 0; i < 2; ++i) {
      int row = wm + i * 16 + lm;
      af[i].q[0] = AsQ[row * 4 + lk];
      af[i].q[1] = AsQ[row * 4 + lk + 2];
    }
    // B fragment: lane holds col N=lm, K = lk*8..+7 and 16+lk*8..+7 (matches A layout)
    FragU bf[4];
#pragma unroll
    for (int j = 0; j < 4; ++j) {
      int nr = wn + j * 16 + lm;
      bf[j].q[0] = BsQ[nr * 4 + lk];
      bf[j].q[1] = BsQ[nr * 4 + lk + 2];
    }
#pragma unroll
    for (int i = 0; i < 2; ++i)
#pragma unroll
      for (int j = 0; j < 4; ++j)
        acc[i][j] = __builtin_amdgcn_wmma_f32_16x16x32_bf16(
            false, af[i].v, false, bf[j].v, (short)0, acc[i][j], false, false);
  }

  // Epilogue: C/D layout -> lane lm = N, VGPR r -> M = lk*8 + r
#pragma unroll
  for (int i = 0; i < 2; ++i) {
#pragma unroll
    for (int j = 0; j < 4; ++j) {
      int n = nbase + wn + j * 16 + lm;
      float badd = 0.f;
      if constexpr (MODE == 0) badd = bias[n];
#pragma unroll
      for (int r = 0; r < 8; ++r) {
        int m = mbase + wm + i * 16 + lk * 8 + r;
        Cout[(size_t)m * ND + n] = acc[i][j][r] + badd;
      }
    }
  }
}

// ---------------------------------------------------------------- LayerNorm
__global__ __launch_bounds__(256) void ln_kernel(const float* __restrict__ cnn,
                                                 const float* __restrict__ g,
                                                 const float* __restrict__ be,
                                                 float* __restrict__ normed,
                                                 __bf16* __restrict__ normed_bf) {
  __shared__ float red[256];
  int m = blockIdx.x, t = threadIdx.x;
  const float4 v = *(const float4*)(cnn + (size_t)m * D_ + t * 4);
  red[t] = v.x + v.y + v.z + v.w;
  __syncthreads();
  for (int o = 128; o > 0; o >>= 1) { if (t < o) red[t] += red[t + o]; __syncthreads(); }
  float mu = red[0] * (1.0f / D_);
  __syncthreads();
  float dx = v.x - mu, dy = v.y - mu, dz = v.z - mu, dw = v.w - mu;
  red[t] = dx * dx + dy * dy + dz * dz + dw * dw;
  __syncthreads();
  for (int o = 128; o > 0; o >>= 1) { if (t < o) red[t] += red[t + o]; __syncthreads(); }
  float rs = rsqrtf(red[0] * (1.0f / D_) + 1e-5f);
  int d0 = t * 4;
  float o0 = dx * rs * g[d0 + 0] + be[d0 + 0];
  float o1 = dy * rs * g[d0 + 1] + be[d0 + 1];
  float o2 = dz * rs * g[d0 + 2] + be[d0 + 2];
  float o3 = dw * rs * g[d0 + 3] + be[d0 + 3];
  *(float4*)(normed + (size_t)m * D_ + d0) = make_float4(o0, o1, o2, o3);
  size_t nb = (size_t)m * D_ + d0;
  normed_bf[nb + 0] = (__bf16)o0; normed_bf[nb + 1] = (__bf16)o1;
  normed_bf[nb + 2] = (__bf16)o2; normed_bf[nb + 3] = (__bf16)o3;
}

// ---------------------------------------------------------------- SRU scan (+lambda mix)
__global__ __launch_bounds__(256) void scan_kernel(const float* __restrict__ U,
                                                   const float* __restrict__ normed,
                                                   const float* __restrict__ cnn,
                                                   const float* __restrict__ sru_v,
                                                   const float* __restrict__ sru_b,
                                                   const float* __restrict__ lambda_w,
                                                   float* __restrict__ out) {
  int g = blockIdx.x * 256 + threadIdx.x;   // 0 .. B*D-1
  int b = g >> 10;
  int d = g & (D_ - 1);
  const float vf = sru_v[d], vr = sru_v[D_ + d];
  const float bfb = sru_b[d], brb = sru_b[D_ + d];
  const float lam = lambda_w[d];
  float c = 0.f;
  for (int l = 0; l < L_; ++l) {
    size_t m = (size_t)l * B_ + b;
    const float* u = U + m * (3 * D_) + 3 * d;
    if (l + 1 < L_)
      __builtin_prefetch(U + (m + B_) * (3 * D_) + 3 * d, 0, 0);
    float u0 = u[0], u1 = u[1], u2 = u[2];
    float xt = normed[m * D_ + d];
    float f  = 1.f / (1.f + __expf(-(u1 + vf * c + bfb)));
    float cn = f * c + (1.f - f) * u0;
    float r  = 1.f / (1.f + __expf(-(u2 + vr * c + brb)));
    float h  = r * tanhf(cn) + (1.f - r) * xt;
    c = cn;
    float co = cnn[m * D_ + d];
    out[((size_t)b * L_ + l) * D_ + d] = lam * co + (1.f - lam) * h;
  }
}

// ---------------------------------------------------------------- RMSNorm (in place on d_out)
__global__ __launch_bounds__(256) void rms_kernel(float* __restrict__ out,
                                                  const float* __restrict__ rw) {
  __shared__ float red[256];
  int row = blockIdx.x, t = threadIdx.x;
  float4 v = *(const float4*)(out + (size_t)row * D_ + t * 4);
  red[t] = v.x * v.x + v.y * v.y + v.z * v.z + v.w * v.w;
  __syncthreads();
  for (int o = 128; o > 0; o >>= 1) { if (t < o) red[t] += red[t + o]; __syncthreads(); }
  float rs = rsqrtf(red[0] * (1.0f / D_) + 1e-6f);
  int d0 = t * 4;
  v.x *= rs * rw[d0 + 0]; v.y *= rs * rw[d0 + 1];
  v.z *= rs * rw[d0 + 2]; v.w *= rs * rw[d0 + 3];
  *(float4*)(out + (size_t)row * D_ + d0) = v;
}

// ---------------------------------------------------------------- launcher
extern "C" void kernel_launch(void* const* d_in, const int* in_sizes, int n_in,
                              void* d_out, int out_size, void* d_ws, size_t ws_size,
                              hipStream_t stream) {
  const float* x        = (const float*)d_in[0];
  const float* conv_w   = (const float*)d_in[1];
  const float* conv_b   = (const float*)d_in[2];
  const float* ln_g     = (const float*)d_in[3];
  const float* ln_b     = (const float*)d_in[4];
  const float* sru_w    = (const float*)d_in[5];
  const float* sru_v    = (const float*)d_in[6];
  const float* sru_b    = (const float*)d_in[7];
  const float* lambda_w = (const float*)d_in[8];
  const float* rms_w    = (const float*)d_in[9];
  float* out = (float*)d_out;

  char* ws = (char*)d_ws;
  size_t off = 0;
  auto alloc = [&](size_t bytes) {
    void* p = ws + off;
    off = (off + bytes + 255) & ~(size_t)255;
    return p;
  };
  uint4*  zpad      = (uint4*)alloc(1024);                        //  1 KB zero pad
  __bf16* x_bf      = (__bf16*)alloc((size_t)B_ * L_ * D_ * 2);   // 32 MB
  __bf16* wc_bf     = (__bf16*)alloc((size_t)D_ * K_ * D_ * 2);   //  6 MB
  __bf16* wsru_bf   = (__bf16*)alloc((size_t)3 * D_ * D_ * 2);    //  6 MB
  float*  cnn       = (float*)alloc((size_t)LB_ * D_ * 4);        // 64 MB
  float*  normed    = (float*)alloc((size_t)LB_ * D_ * 4);        // 64 MB
  __bf16* normed_bf = (__bf16*)alloc((size_t)LB_ * D_ * 2);       // 32 MB
  float*  U         = (float*)alloc((size_t)LB_ * 3 * D_ * 4);    // 192 MB

  pack_x_kernel<<<(B_ * L_ * D_) / 1024, 256, 0, stream>>>(x, x_bf, zpad);
  pack_wc_kernel<<<(D_ * K_ * D_) / 256, 256, 0, stream>>>(conv_w, wc_bf);
  pack_ws_kernel<<<(3 * D_ * D_) / 256, 256, 0, stream>>>(sru_w, wsru_bf);

  gemm_wmma_kernel<0><<<dim3(D_ / 128, LB_ / 128), 256, 0, stream>>>(
      x_bf, wc_bf, conv_b, (const __bf16*)zpad, cnn);
  ln_kernel<<<LB_, 256, 0, stream>>>(cnn, ln_g, ln_b, normed, normed_bf);
  gemm_wmma_kernel<1><<<dim3((3 * D_) / 128, LB_ / 128), 256, 0, stream>>>(
      normed_bf, wsru_bf, nullptr, nullptr, U);
  scan_kernel<<<(B_ * D_) / 256, 256, 0, stream>>>(U, normed, cnn, sru_v, sru_b, lambda_w, out);
  rms_kernel<<<B_ * L_, 256, 0, stream>>>(out, rms_w);
}
